// NLSRM_55250459295984
// MI455X (gfx1250) — compile-verified, matrix operand order checked by reading, not run on hardware
//
#include <hip/hip_runtime.h>

// NLSRM non-local + SRM block for MI455X (gfx1250).
// Big GEMMs (attention QK^T and P*V, 34 of 39 GFLOP) use bf16x3 error-
// compensated V_WMMA_F32_16X16X32_BF16 (hi*hi + hi*lo + lo*hi, fp32 acc,
// ~1e-5 rel err). Small GEMMs (QKV, out-proj) stay exact f32 WMMA.
// Layouts (all fragment loads are single aligned b64/b128):
//   s      [N][C][L]  fp32   (residual + QKV B operand)
//   th/tl  [N][L][Ch] bf16   (theta hi/lo, transposed)
//   ph/pl  [N][L][Ch] bf16   (phi hi/lo, transposed)
//   gh/gl  [N][Ch][L] bf16   (g hi/lo)
//   nl_t   [N][L][Ch] fp32
// Workspace total: 48 MB.

typedef __attribute__((ext_vector_type(2)))  float    v2f;
typedef __attribute__((ext_vector_type(8)))  float    v8f;
typedef __attribute__((ext_vector_type(8)))  __bf16   v8bf;
typedef __attribute__((ext_vector_type(16))) __bf16   v16bf;
typedef __attribute__((ext_vector_type(4)))  unsigned v4u;

#define CC   256
#define CH   128
#define LL   4096
#define NB   4
#define EPSV 1e-5f
#define SROW 4100   // LDS row stride (words): 4100 % 64 == 4 -> conflict-free

__device__ __forceinline__ v8f wmma_f32(v2f a, v2f b, v8f c) {
  return __builtin_amdgcn_wmma_f32_16x16x4_f32(false, a, false, b, (short)0, c,
                                               false, false);
}
__device__ __forceinline__ v8f wmma_bf16(v16bf a, v16bf b, v8f c) {
  return __builtin_amdgcn_wmma_f32_16x16x32_bf16(false, a, false, b, (short)0, c,
                                                 false, false);
}
__device__ __forceinline__ v2f ld2(const float* p) { return *(const v2f*)p; }

// Load one 16-element bf16 fragment: two contiguous 16B runs 16 bf16 apart
// (matches the 16-bit A/B per-lane K layout: [8h,8h+8) and [8h+16,8h+24)).
__device__ __forceinline__ v16bf ldfrag(const __bf16* p) {
  v8bf a = *(const v8bf*)p;
  v8bf b = *(const v8bf*)(p + 16);
  return __builtin_shufflevector(a, b, 0, 1, 2, 3, 4, 5, 6, 7,
                                 8, 9, 10, 11, 12, 13, 14, 15);
}
__device__ __forceinline__ __bf16 bfhi(unsigned w) {
  return __builtin_bit_cast(__bf16, (unsigned short)(w >> 16));
}
__device__ __forceinline__ __bf16 bflo(unsigned w) {
  return __builtin_bit_cast(__bf16, (unsigned short)(w & 0xffffu));
}

// ---------------------------------------------------------------------------
// Kernel 1: SRM pooling + gate + s = x * sigmoid(mean*cfc0 + std*cfc1)
// ---------------------------------------------------------------------------
__global__ __launch_bounds__(256)
void nlsrm_srm_kernel(const float* __restrict__ x, const float* __restrict__ cfc,
                      float* __restrict__ s) {
  const int nc = blockIdx.x;                 // n*C + c
  const int c  = nc & (CC - 1);
  const float4* xv = (const float4*)(x + (size_t)nc * LL);
  float4*       sv = (float4*)(s + (size_t)nc * LL);
  const int t = threadIdx.x;

  float sum = 0.f, sq = 0.f;
#pragma unroll
  for (int k = 0; k < 4; ++k) {
    float4 v = xv[t + 256 * k];
    sum += v.x + v.y + v.z + v.w;
    sq  += v.x * v.x + v.y * v.y + v.z * v.z + v.w * v.w;
  }

  __shared__ float r0[256], r1[256];
  __shared__ float gsh;
  r0[t] = sum; r1[t] = sq;
  __syncthreads();
  for (int off = 128; off > 0; off >>= 1) {
    if (t < off) { r0[t] += r0[t + off]; r1[t] += r1[t + off]; }
    __syncthreads();
  }
  if (t == 0) {
    float mean = r0[0] * (1.f / (float)LL);
    float var  = (r1[0] - (float)LL * mean * mean) * (1.f / (float)(LL - 1)) + EPSV;
    float sd   = sqrtf(var);
    float z    = mean * cfc[c * 2 + 0] + sd * cfc[c * 2 + 1];
    gsh        = 1.f / (1.f + __expf(-z));
  }
  __syncthreads();
  const float gate = gsh;
#pragma unroll
  for (int k = 0; k < 4; ++k) {
    float4 v = xv[t + 256 * k];
    v.x *= gate; v.y *= gate; v.z *= gate; v.w *= gate;
    sv[t + 256 * k] = v;
  }
}

// ---------------------------------------------------------------------------
// Kernel 2: fused QKV GEMM (exact f32 WMMA), emitting bf16 hi/lo splits.
// One B-fragment load feeds 3 WMMAs. theta/phi stored transposed [L][Ch].
// ---------------------------------------------------------------------------
__global__ __launch_bounds__(256)
void nlsrm_qkv_kernel(const float* __restrict__ s,
                      const float* __restrict__ tw, const float* __restrict__ tb,
                      const float* __restrict__ pw, const float* __restrict__ pb,
                      const float* __restrict__ gw, const float* __restrict__ gb,
                      __bf16* __restrict__ thh, __bf16* __restrict__ thl,
                      __bf16* __restrict__ phh, __bf16* __restrict__ phl,
                      __bf16* __restrict__ ghp, __bf16* __restrict__ glp) {
  const int j0 = blockIdx.x * 16;
  const int n  = blockIdx.y;
  const float* sp = s + (size_t)n * CC * LL;
  const size_t toff = (size_t)n * LL * CH;   // theta/phi transposed base
  const size_t goff = (size_t)n * CH * LL;

  const int lane = threadIdx.x & 31;
  const int wave = threadIdx.x >> 5;
  const int m    = lane & 15;
  const int kb   = (lane >> 4) * 2;
  const int half = lane >> 4;
  const int m0   = wave * 16;

  v8f at = {0.f,0.f,0.f,0.f,0.f,0.f,0.f,0.f};
  v8f ap = at, ag = at;
#pragma unroll
  for (int t = 0; t < CC / 4; ++t) {
    const int k0 = 4 * t + kb;
    v2f b;
    b.x = sp[(size_t)(k0)     * LL + j0 + m];
    b.y = sp[(size_t)(k0 + 1) * LL + j0 + m];
    at = wmma_f32(ld2(tw + (m0 + m) * CC + k0), b, at);
    ap = wmma_f32(ld2(pw + (m0 + m) * CC + k0), b, ap);
    ag = wmma_f32(ld2(gw + (m0 + m) * CC + k0), b, ag);
  }
#pragma unroll
  for (int r = 0; r < 8; ++r) {
    const int mr = m0 + r + 8 * half;
    const size_t ti = toff + (size_t)(j0 + m) * CH + mr;  // consecutive in r
    const size_t gi = goff + (size_t)mr * LL + j0 + m;
    float vt = at[r] + tb[mr];
    __bf16 ht = (__bf16)vt;
    thh[ti] = ht; thl[ti] = (__bf16)(vt - (float)ht);
    float vp = ap[r] + pb[mr];
    __bf16 hp = (__bf16)vp;
    phh[ti] = hp; phl[ti] = (__bf16)(vp - (float)hp);
    float vg = ag[r] + gb[mr];
    __bf16 hg = (__bf16)vg;
    ghp[gi] = hg; glp[gi] = (__bf16)(vg - (float)hg);
  }
}

// ---------------------------------------------------------------------------
// Kernel 3: fused attention in the 320KB WGP LDS, bf16x3 WMMA.
// One block per (n, 16-row query tile). Padded 16x4096 fp32 logits stripe
// -> softmax in place -> repack P as {hi16,lo16} words -> O = P @ g^T.
// ---------------------------------------------------------------------------
__global__ __launch_bounds__(256)
void nlsrm_attn_kernel(const __bf16* __restrict__ thh, const __bf16* __restrict__ thl,
                       const __bf16* __restrict__ phh, const __bf16* __restrict__ phl,
                       const __bf16* __restrict__ ghp, const __bf16* __restrict__ glp,
                       float* __restrict__ nlt) {
  __shared__ float Slds[16 * SROW];    // ~262 KB, padded rows
  __shared__ float red[256];
  __shared__ float rowstat[16];

  const int i0 = blockIdx.x * 16;
  const int n  = blockIdx.y;
  const size_t toff = (size_t)n * LL * CH;
  const size_t goff = (size_t)n * CH * LL;

  const int lane = threadIdx.x & 31;
  const int wave = threadIdx.x >> 5;
  const int m    = lane & 15;
  const int half = lane >> 4;

  // Preload Q hi/lo fragments: 4 chunks of K=32 from theta_t row (i0+m).
  const __bf16* qhr = thh + toff + (size_t)(i0 + m) * CH + 8 * half;
  const __bf16* qlr = thl + toff + (size_t)(i0 + m) * CH + 8 * half;
  v16bf qh[4], ql[4];
#pragma unroll
  for (int ck = 0; ck < 4; ++ck) {
    qh[ck] = ldfrag(qhr + 32 * ck);
    ql[ck] = ldfrag(qlr + 32 * ck);
  }

  // ---- Phase 1: logits S[16, 4096] into LDS (waves split j-tiles) ----
  for (int jt = wave; jt < LL / 16; jt += 8) {
    const int j0 = jt * 16;
    const __bf16* bhr = phh + toff + (size_t)(j0 + m) * CH + 8 * half;
    const __bf16* blr = phl + toff + (size_t)(j0 + m) * CH + 8 * half;
    v8f acc = {0.f,0.f,0.f,0.f,0.f,0.f,0.f,0.f};
#pragma unroll
    for (int ck = 0; ck < 4; ++ck) {
      v16bf bh = ldfrag(bhr + 32 * ck);
      v16bf bl = ldfrag(blr + 32 * ck);
      acc = wmma_bf16(qh[ck], bh, acc);   // hi*hi
      acc = wmma_bf16(qh[ck], bl, acc);   // hi*lo
      acc = wmma_bf16(ql[ck], bh, acc);   // lo*hi
    }
#pragma unroll
    for (int r = 0; r < 8; ++r)
      Slds[(r + 8 * half) * SROW + j0 + m] = acc[r];
  }
  __syncthreads();

  // ---- Phase 2: row softmax in LDS; final pass packs P as {hi16,lo16} ----
  const int row = threadIdx.x >> 4;
  const int sub = threadIdx.x & 15;
  float* Srow = Slds + row * SROW;

  float mx = -3.4e38f;
  for (int k = sub; k < LL; k += 16) mx = fmaxf(mx, Srow[k]);
  red[threadIdx.x] = mx;
  __syncthreads();
  if (sub == 0) {
    float v = red[row * 16];
    for (int q = 1; q < 16; ++q) v = fmaxf(v, red[row * 16 + q]);
    rowstat[row] = v;
  }
  __syncthreads();
  const float rmax = rowstat[row];
  float sm = 0.f;
  for (int k = sub; k < LL; k += 16) {
    float p = __expf(Srow[k] - rmax);
    Srow[k] = p;
    sm += p;
  }
  __syncthreads();
  red[threadIdx.x] = sm;
  __syncthreads();
  if (sub == 0) {
    float v = 0.f;
    for (int q = 0; q < 16; ++q) v += red[row * 16 + q];
    rowstat[row] = v;
  }
  __syncthreads();
  const float rinv = 1.f / rowstat[row];
  unsigned* Urow = (unsigned*)Srow;
  for (int k = sub; k < LL; k += 16) {
    float p = Srow[k] * rinv;
    __bf16 h = (__bf16)p;
    __bf16 l = (__bf16)(p - (float)h);
    Urow[k] = ((unsigned)__builtin_bit_cast(unsigned short, h) << 16) |
              (unsigned)__builtin_bit_cast(unsigned short, l);
  }
  __syncthreads();

  // ---- Phase 3: O[i, c] = sum_j P[i,j] * g[c,j]; wave w -> c-tile 16w ----
  const int c0 = wave * 16;
  const unsigned* Pm  = (const unsigned*)Slds + m * SROW + 8 * half;
  const __bf16*   ghr = ghp + goff + (size_t)(c0 + m) * LL + 8 * half;
  const __bf16*   glr = glp + goff + (size_t)(c0 + m) * LL + 8 * half;
  v8f acc = {0.f,0.f,0.f,0.f,0.f,0.f,0.f,0.f};
#pragma unroll 4
  for (int ck = 0; ck < LL / 32; ++ck) {
    // A fragment: unpack 16 packed words (two b128 pairs) into hi/lo bf16.
    v4u w0 = *(const v4u*)(Pm + 32 * ck);
    v4u w1 = *(const v4u*)(Pm + 32 * ck + 4);
    v4u w2 = *(const v4u*)(Pm + 32 * ck + 16);
    v4u w3 = *(const v4u*)(Pm + 32 * ck + 20);
    v16bf ah, al;
#pragma unroll
    for (int e = 0; e < 4; ++e) {
      ah[e]      = bfhi(w0[e]); al[e]      = bflo(w0[e]);
      ah[e + 4]  = bfhi(w1[e]); al[e + 4]  = bflo(w1[e]);
      ah[e + 8]  = bfhi(w2[e]); al[e + 8]  = bflo(w2[e]);
      ah[e + 12] = bfhi(w3[e]); al[e + 12] = bflo(w3[e]);
    }
    v16bf bh = ldfrag(ghr + 32 * ck);
    v16bf bl = ldfrag(glr + 32 * ck);
    acc = wmma_bf16(ah, bh, acc);
    acc = wmma_bf16(ah, bl, acc);
    acc = wmma_bf16(al, bh, acc);
  }
  float* np = nlt + (size_t)n * LL * CH;
#pragma unroll
  for (int r = 0; r < 8; ++r)
    np[(size_t)(i0 + r + 8 * half) * CH + c0 + m] = acc[r];
}

// ---------------------------------------------------------------------------
// Kernel 4: out = s + out_w[256,128] @ nl[128,L] + out_b (exact f32 WMMA).
// ---------------------------------------------------------------------------
__global__ __launch_bounds__(256)
void nlsrm_out_kernel(const float* __restrict__ s, const float* __restrict__ nlt,
                      const float* __restrict__ ow, const float* __restrict__ ob,
                      float* __restrict__ out) {
  const int j0 = blockIdx.x * 16;
  const int n  = blockIdx.y;
  const float* np = nlt + (size_t)n * LL * CH;
  const float* sp = s   + (size_t)n * CC * LL;
  float*       op = out + (size_t)n * CC * LL;

  const int lane = threadIdx.x & 31;
  const int wave = threadIdx.x >> 5;
  const int m    = lane & 15;
  const int kb   = (lane >> 4) * 2;
  const int half = lane >> 4;
  const int m0   = wave * 16;           // tile 0 rows; tile 1 = m0 + 128

  const float* brow = np + (size_t)(j0 + m) * CH + kb;
  const float* a0r  = ow + (m0 + m) * CH + kb;
  const float* a1r  = ow + (m0 + 128 + m) * CH + kb;

  v8f acc0 = {0.f,0.f,0.f,0.f,0.f,0.f,0.f,0.f};
  v8f acc1 = acc0;
#pragma unroll
  for (int t = 0; t < CH / 4; ++t) {
    v2f b = ld2(brow + 4 * t);
    acc0 = wmma_f32(ld2(a0r + 4 * t), b, acc0);
    acc1 = wmma_f32(ld2(a1r + 4 * t), b, acc1);
  }
#pragma unroll
  for (int r = 0; r < 8; ++r) {
    const int mr0 = m0 + r + 8 * half;
    const size_t i0x = (size_t)mr0 * LL + j0 + m;
    op[i0x] = sp[i0x] + acc0[r] + ob[mr0];
    const int mr1 = mr0 + 128;
    const size_t i1x = (size_t)mr1 * LL + j0 + m;
    op[i1x] = sp[i1x] + acc1[r] + ob[mr1];
  }
}

// ---------------------------------------------------------------------------
extern "C" void kernel_launch(void* const* d_in, const int* in_sizes, int n_in,
                              void* d_out, int out_size, void* d_ws, size_t ws_size,
                              hipStream_t stream) {
  const float* x   = (const float*)d_in[0];
  const float* cfc = (const float*)d_in[1];
  const float* tw  = (const float*)d_in[2];
  const float* tb  = (const float*)d_in[3];
  const float* pw  = (const float*)d_in[4];
  const float* pb  = (const float*)d_in[5];
  const float* gw  = (const float*)d_in[6];
  const float* gb  = (const float*)d_in[7];
  const float* ow  = (const float*)d_in[8];
  const float* ob  = (const float*)d_in[9];
  float* out = (float*)d_out;

  char* w = (char*)d_ws;
  const size_t S_BYTES  = (size_t)NB * CC * LL * 4;   // 16 MB fp32
  const size_t TP_BYTES = (size_t)NB * LL * CH * 2;   // 4 MB bf16 each
  float*  s   = (float*)(w);
  __bf16* thh = (__bf16*)(w + S_BYTES);
  __bf16* thl = (__bf16*)(w + S_BYTES + TP_BYTES);
  __bf16* phh = (__bf16*)(w + S_BYTES + 2 * TP_BYTES);
  __bf16* phl = (__bf16*)(w + S_BYTES + 3 * TP_BYTES);
  __bf16* ghp = (__bf16*)(w + S_BYTES + 4 * TP_BYTES);
  __bf16* glp = (__bf16*)(w + S_BYTES + 5 * TP_BYTES);
  float*  nlt = (float*) (w + S_BYTES + 6 * TP_BYTES); // 8 MB fp32

  nlsrm_srm_kernel<<<NB * CC, 256, 0, stream>>>(x, cfc, s);
  nlsrm_qkv_kernel<<<dim3(LL / 16, NB), 256, 0, stream>>>(
      s, tw, tb, pw, pb, gw, gb, thh, thl, phh, phl, ghp, glp);
  nlsrm_attn_kernel<<<dim3(LL / 16, NB), 256, 0, stream>>>(
      thh, thl, phh, phl, ghp, glp, nlt);
  nlsrm_out_kernel<<<dim3(LL / 16, NB), 256, 0, stream>>>(s, nlt, ow, ob, out);
}
